// BahdanauAttention_49778670961165
// MI455X (gfx1250) — matrix operand order repaired and checked by reading, not compiled, not run on hardware
//
#include <hip/hip_runtime.h>
#include <hip/hip_bf16.h>
#include <math.h>

// Problem constants (match reference)
#define B_      32
#define T_      4096
#define D_      256
#define U_      256
#define TILE_T  128   // rows of T per block in the score kernel (8 waves x 16 rows)

typedef __attribute__((ext_vector_type(16))) __bf16        v16bf;
typedef __attribute__((ext_vector_type(8)))  float         v8f;
typedef __attribute__((ext_vector_type(4))) unsigned int   u32x4;
typedef __attribute__((ext_vector_type(8)))  int           i32x8;
typedef __attribute__((ext_vector_type(4)))  int           i32x4;
typedef __attribute__((ext_vector_type(4)))  float         f32x4;

#if defined(__has_builtin)
#if __has_builtin(__builtin_amdgcn_tensor_load_to_lds)
#define HAVE_TDM 1
#endif
#if __has_builtin(__builtin_amdgcn_tanhf)
#define FAST_TANH(x) __builtin_amdgcn_tanhf(x)   // v_tanh_f32 (TRANS32)
#endif
#endif
#ifndef FAST_TANH
#define FAST_TANH(x) tanhf(x)
#endif

// ---------------------------------------------------------------------------
// Prep kernel 1: W1 [D,U] f32 (row-major) -> W1T [U,D] bf16
// so that WMMA B-fragments are contiguous 16x bf16 loads per lane.
// ---------------------------------------------------------------------------
__global__ __launch_bounds__(256)
void w1t_bf16_kernel(const float* __restrict__ W1, __bf16* __restrict__ W1T)
{
    const int u = blockIdx.x;   // 0..U-1
    const int d = threadIdx.x;  // 0..D-1
    W1T[(size_t)u * D_ + d] = (__bf16)W1[(size_t)d * U_ + u];
}

// ---------------------------------------------------------------------------
// Prep kernel 2: bias[b,u] = query[b,:] @ W2[:,u] + b2[u] + b1[u]
// (b1 folded in so the score kernel's epilogue does one load, not two)
// ---------------------------------------------------------------------------
__global__ __launch_bounds__(256)
void projq_kernel(const float* __restrict__ query,
                  const float* __restrict__ W2,
                  const float* __restrict__ b2,
                  const float* __restrict__ b1,
                  float* __restrict__ bias)
{
    const int b = blockIdx.x;
    const int u = threadIdx.x;
    float s = b2[u] + b1[u];
#pragma unroll 4
    for (int d = 0; d < D_; ++d)
        s = fmaf(query[b * D_ + d], W2[(size_t)d * U_ + u], s);
    bias[b * U_ + u] = s;
}

// ---------------------------------------------------------------------------
// Main kernel: raw scores[b,t] = tanh(values[b,t,:]@W1 + bias[b,:]) @ V
// grid = (T/TILE_T, B); 256 threads = 8 waves; wave w owns rows w*16..w*16+15.
// values tile (TILE_T x D f32, 128 KB) staged into LDS by the Tensor Data
// Mover; W1T stays L2-hot (128 KB total, reused by all 1024 blocks).
// ---------------------------------------------------------------------------
__global__ __launch_bounds__(256)
void bahdanau_score_kernel(const float* __restrict__ values,
                           const __bf16* __restrict__ W1T,
                           const float* __restrict__ bias,
                           const float* __restrict__ Vw,
                           float* __restrict__ scoreOut)
{
    extern __shared__ float tileV[];          // TILE_T * D_ floats (128 KB)

    const int b    = blockIdx.y;
    const int t0   = blockIdx.x * TILE_T;
    const int tid  = threadIdx.x;
    const int lane = tid & 31;
    const int wave = tid >> 5;                // 0..7
    const int mRow = lane & 15;               // row-within-16 / N column
    const int halfK = lane >> 4;              // lane half selects K sub-block
    const int rowBase = wave * 16;            // rows of tile owned by this wave

    // ---- stage values tile into LDS --------------------------------------
#if HAVE_TDM
    if (wave == 0) {
        // D# per CDNA5 ISA ch.8: 2-D tensor, data_size=4B,
        // tile = 256 x 128, tensor_dim0_stride = 256 elements.
        const unsigned long long ga =
            (unsigned long long)(const void*)(values + ((size_t)b * T_ + t0) * D_);
        const unsigned ldsOff = (unsigned)(unsigned long long)(void*)&tileV[0];

        u32x4 g0;
        g0[0] = 1u;                                            // count=1 (valid), user mode
        g0[1] = ldsOff;                                        // lds_addr (bytes)
        g0[2] = (unsigned)(ga & 0xFFFFFFFFull);                // global_addr[31:0]
        g0[3] = (unsigned)((ga >> 32) & 0x01FFFFFFull)         // global_addr[56:32]
              | 0x80000000u;                                   // type=2 ("image")

        i32x8 g1;
        g1[0] = 0x00020000;                                    // data_size=2 -> 4 bytes
        g1[1] = (int)(((unsigned)D_ & 0xFFFFu) << 16);         // tensor_dim0 lo16
        g1[2] = (int)((((unsigned)D_ >> 16) & 0xFFFFu)         // tensor_dim0 hi16
              | (((unsigned)TILE_T & 0xFFFFu) << 16));         // tensor_dim1 lo16
        g1[3] = (int)((((unsigned)TILE_T >> 16) & 0xFFFFu)     // tensor_dim1 hi16
              | ((unsigned)D_ << 16));                         // tile_dim0 = 256
        g1[4] = (int)((unsigned)TILE_T);                       // tile_dim1=128, tile_dim2=0
        g1[5] = (int)((unsigned)D_);                           // tensor_dim0_stride lo32
        g1[6] = 0;                                             // stride hi / dim1_stride lo
        g1[7] = 0;

        i32x4 g2 = {0, 0, 0, 0};                               // unused (2-D tensor)
        i32x4 g3 = {0, 0, 0, 0};
        i32x8 g4 = {0, 0, 0, 0, 0, 0, 0, 0};                   // 6-arg toolchain extra group

        __builtin_amdgcn_tensor_load_to_lds(g0, g1, g2, g3, g4, 0);
        __builtin_amdgcn_s_wait_tensorcnt(0);
    }
#else
    {
        const f32x4* src = (const f32x4*)(values + ((size_t)b * T_ + t0) * D_);
        f32x4* dst = (f32x4*)tileV;
        for (int i = tid; i < TILE_T * D_ / 4; i += 256) dst[i] = src[i];
    }
#endif
    __syncthreads();

    // ---- build the 8 A-fragments (K tiles of 32) once, f32->bf16 ---------
    // A 16x32 in this code's contiguous k order (B matches): lanes 0-15 hold K 0-15, lanes 16-31 hold K 16-31.
    v16bf aF[8];
    {
        const float* rp = &tileV[(rowBase + mRow) * D_];
#pragma unroll
        for (int kt = 0; kt < 8; ++kt) {
            const float* s0 = rp + kt * 32 + halfK * 16;
#pragma unroll
            for (int j = 0; j < 8; ++j) {
                aF[kt][j]     = (__bf16)s0[j];
                aF[kt][j + 8] = (__bf16)s0[j + 8];
            }
        }
    }

    float sc[8];
#pragma unroll
    for (int r = 0; r < 8; ++r) sc[r] = 0.0f;

    // ---- sweep the 16 N-tiles of U; epilogue fuses bias+tanh+dot(V) ------
#pragma unroll 1
    for (int nt = 0; nt < 16; ++nt) {
        const int u = nt * 16 + mRow;             // N column this lane owns in C
        v8f acc = {0.f, 0.f, 0.f, 0.f, 0.f, 0.f, 0.f, 0.f};

        const __bf16* wp = &W1T[(size_t)u * D_ + halfK * 16];
#pragma unroll
        for (int kt = 0; kt < 8; ++kt) {
            const v16bf bF = *(const v16bf*)(wp + kt * 32);
            acc = __builtin_amdgcn_wmma_f32_16x16x32_bf16(
                      false, aF[kt], false, bF, (short)0, acc, false, false);
        }

        const float bs = bias[b * U_ + u];
        const float vu = Vw[u];
#pragma unroll
        for (int r = 0; r < 8; ++r)
            sc[r] += FAST_TANH(acc[r] + bs) * vu; // C: VGPR r -> M = r + 8*halfK
    }

    // ---- reduce across the 16 N-lanes of each half-wave ------------------
#pragma unroll
    for (int off = 1; off < 16; off <<= 1)
#pragma unroll
        for (int r = 0; r < 8; ++r)
            sc[r] += __shfl_xor(sc[r], off, 32);

    if (mRow == 0) {
        float* out = &scoreOut[(size_t)b * T_ + t0 + rowBase + halfK * 8];
#pragma unroll
        for (int r = 0; r < 8; ++r) out[r] = sc[r];
    }
}

// ---------------------------------------------------------------------------
// Softmax over T per batch, in place on the attn region of d_out.
// 256 threads x 16 values/thread, register-blocked; wave + LDS reductions.
// ---------------------------------------------------------------------------
__global__ __launch_bounds__(256)
void softmax_kernel(float* __restrict__ attn)
{
    const int b    = blockIdx.x;
    const int tid  = threadIdx.x;
    const int lane = tid & 31;
    const int wave = tid >> 5;
    float* base = attn + (size_t)b * T_;

    float v[16];
    float m = -__builtin_inff();
#pragma unroll
    for (int i = 0; i < 16; ++i) {
        v[i] = base[tid + i * 256];
        m = fmaxf(m, v[i]);
    }
#pragma unroll
    for (int off = 16; off > 0; off >>= 1) m = fmaxf(m, __shfl_xor(m, off, 32));

    __shared__ float redM[8];
    __shared__ float redS[8];
    if (lane == 0) redM[wave] = m;
    __syncthreads();
    if (wave == 0) {
        float t = (lane < 8) ? redM[lane] : -__builtin_inff();
#pragma unroll
        for (int off = 4; off > 0; off >>= 1) t = fmaxf(t, __shfl_xor(t, off, 32));
        if (lane == 0) redM[0] = t;
    }
    __syncthreads();
    m = redM[0];

    float s = 0.0f;
#pragma unroll
    for (int i = 0; i < 16; ++i) {
        v[i] = __expf(v[i] - m);
        s += v[i];
    }
#pragma unroll
    for (int off = 16; off > 0; off >>= 1) s += __shfl_xor(s, off, 32);
    if (lane == 0) redS[wave] = s;
    __syncthreads();
    if (wave == 0) {
        float t = (lane < 8) ? redS[lane] : 0.0f;
#pragma unroll
        for (int off = 4; off > 0; off >>= 1) t += __shfl_xor(t, off, 32);
        if (lane == 0) redS[0] = t;
    }
    __syncthreads();

    const float inv = 1.0f / redS[0];
#pragma unroll
    for (int i = 0; i < 16; ++i) base[tid + i * 256] = v[i] * inv;
}

// ---------------------------------------------------------------------------
// context[b,d] = sum_t attn[b,t] * values[b,t,d]  (values L2-resident by now)
// ---------------------------------------------------------------------------
__global__ __launch_bounds__(256)
void context_kernel(const float* __restrict__ values,
                    const float* __restrict__ attn,
                    float* __restrict__ ctx)
{
    const int b = blockIdx.x;
    const int d = threadIdx.x;
    __shared__ float sa[256];

    float acc = 0.0f;
    for (int t0 = 0; t0 < T_; t0 += 256) {
        __syncthreads();
        sa[threadIdx.x] = attn[(size_t)b * T_ + t0 + threadIdx.x];
        __syncthreads();
        const float* vp = values + ((size_t)b * T_ + t0) * D_ + d;
        if (t0 + 256 < T_)
            __builtin_prefetch(values + ((size_t)b * T_ + t0 + 256) * D_ + d, 0, 1);
#pragma unroll 8
        for (int j = 0; j < 256; ++j)
            acc = fmaf(sa[j], vp[(size_t)j * D_], acc);
    }
    ctx[(size_t)b * D_ + d] = acc;
}

// ---------------------------------------------------------------------------
// Host-side launcher
// ---------------------------------------------------------------------------
extern "C" void kernel_launch(void* const* d_in, const int* in_sizes, int n_in,
                              void* d_out, int out_size, void* d_ws, size_t ws_size,
                              hipStream_t stream)
{
    (void)in_sizes; (void)n_in; (void)out_size; (void)ws_size;

    const float* query  = (const float*)d_in[0];   // [B,D]
    const float* values = (const float*)d_in[1];   // [B,T,D]
    const float* W1     = (const float*)d_in[2];   // [D,U]
    const float* b1     = (const float*)d_in[3];   // [U]
    const float* W2     = (const float*)d_in[4];   // [D,U]
    const float* b2     = (const float*)d_in[5];   // [U]
    const float* Vw     = (const float*)d_in[6];   // [U,1]
    // d_in[7] = bV: cancels in softmax, does not affect attn or context.

    float* ctx  = (float*)d_out;                       // [B,D]
    float* attn = (float*)d_out + (size_t)B_ * D_;     // [B,T] (scores -> softmax in place)

    __bf16* W1T  = (__bf16*)d_ws;                                  // 128 KB
    float*  bias = (float*)((char*)d_ws + (size_t)U_ * D_ * 2);    //  32 KB

    w1t_bf16_kernel<<<U_, D_, 0, stream>>>(W1, W1T);
    projq_kernel<<<B_, U_, 0, stream>>>(query, W2, b2, b1, bias);

    dim3 grid(T_ / TILE_T, B_);
    bahdanau_score_kernel<<<grid, 256, TILE_T * D_ * sizeof(float), stream>>>(
        values, W1T, bias, Vw, attn);

    softmax_kernel<<<B_, 256, 0, stream>>>(attn);
    context_kernel<<<B_, D_, 0, stream>>>(values, attn, ctx);
}
